// DDINOLoss_2937757630884
// MI455X (gfx1250) — compile-verified, matrix operand order checked by reading, not run on hardware
//
#include <hip/hip_runtime.h>
#include <hip/hip_bf16.h>

// ---------------------------------------------------------------------------
// DINO-style distillation loss for MI455X (gfx1250, wave32, WMMA).
//
// Pipeline (all on `stream`):
//  K0  zero scalar accumulator
//  K1  teacher cls softmax  q = softmax((t_cls - center)/temp)      [64 x 4096]
//  K2  teacher region stats (rowmax, rowsumexp) of (t_reg-cg)/temp  [12544 x 2]
//  K3  row LSE of student_region/0.1 and student_cls/0.1            (float4)
//  K4  L2-normalize feats -> f16 (WMMA operand precision)
//  K5  per-pair sim = s_fea . t_fea^T via v_wmma_f32_16x16x32_f16,
//      per-lane running argmax, single end-of-wave butterfly
//  K6  per-pair fused CE: lse - dot(softmax_teacher_row, s_row*10)  (float4)
//  K7  finalize: d_out[0] = acc
// ---------------------------------------------------------------------------

typedef __attribute__((ext_vector_type(16))) _Float16 v16h;
typedef __attribute__((ext_vector_type(8)))  _Float16 v8h;
typedef __attribute__((ext_vector_type(4)))  _Float16 v4h;
typedef __attribute__((ext_vector_type(8)))  float    v8f;

#define DD 4096   // OUT_DIM
#define FF 384    // DFEAT
#define NEGINF (-3.4e38f)

__device__ __forceinline__ float temp_from_epoch(int epoch) {
    // linspace(0.04, 0.07, 30) for warmup epochs, then 0.07
    if (epoch >= 30) return 0.07f;
    if (epoch < 0) epoch = 0;
    return 0.04f + (0.07f - 0.04f) * ((float)epoch * (1.0f / 29.0f));
}

__device__ __forceinline__ float block_sum(float v, float* sm) {
    int t = threadIdx.x;
    sm[t] = v; __syncthreads();
    for (int s = blockDim.x >> 1; s > 0; s >>= 1) {
        if (t < s) sm[t] += sm[t + s];
        __syncthreads();
    }
    float r = sm[0]; __syncthreads();
    return r;
}

__device__ __forceinline__ float block_max(float v, float* sm) {
    int t = threadIdx.x;
    sm[t] = v; __syncthreads();
    for (int s = blockDim.x >> 1; s > 0; s >>= 1) {
        if (t < s) sm[t] = fmaxf(sm[t], sm[t + s]);
        __syncthreads();
    }
    float r = sm[0]; __syncthreads();
    return r;
}

// ---------------------------------------------------------------- K0 / K7
__global__ void k_zero(float* acc) { if (threadIdx.x == 0) acc[0] = 0.0f; }
__global__ void k_finalize(const float* acc, float* out) {
    if (threadIdx.x == 0) out[0] = acc[0];
}

// ---------------------------------------------------------------- K1
// q[row] = softmax((teacher_cls[row] - center) / temp)   (float4 streaming)
__global__ void k_teacher_cls_softmax(const float* __restrict__ tcls,
                                      const float* __restrict__ center,
                                      const int* __restrict__ epoch,
                                      float* __restrict__ q) {
    __shared__ float sm[256];
    const int row = blockIdx.x;
    const float invt = 1.0f / temp_from_epoch(*epoch);
    const float4* x4 = (const float4*)(tcls + (size_t)row * DD);
    const float4* c4 = (const float4*)center;
    float4*       q4 = (float4*)(q + (size_t)row * DD);

    float m = NEGINF;
    for (int d = threadIdx.x; d < DD / 4; d += blockDim.x) {
        float4 x = x4[d], c = c4[d];
        m = fmaxf(m, fmaxf(fmaxf((x.x - c.x), (x.y - c.y)),
                           fmaxf((x.z - c.z), (x.w - c.w))) * invt);
        m = fmaxf(m, fminf(fminf((x.x - c.x), (x.y - c.y)),
                           fminf((x.z - c.z), (x.w - c.w))) * invt);  // invt>0, but keep robust
    }
    m = block_max(m, sm);
    float s = 0.0f;
    for (int d = threadIdx.x; d < DD / 4; d += blockDim.x) {
        float4 x = x4[d], c = c4[d];
        s += __expf((x.x - c.x) * invt - m) + __expf((x.y - c.y) * invt - m)
           + __expf((x.z - c.z) * invt - m) + __expf((x.w - c.w) * invt - m);
    }
    s = block_sum(s, sm);
    const float invZ = 1.0f / s;
    for (int d = threadIdx.x; d < DD / 4; d += blockDim.x) {
        float4 x = x4[d], c = c4[d], o;
        o.x = __expf((x.x - c.x) * invt - m) * invZ;
        o.y = __expf((x.y - c.y) * invt - m) * invZ;
        o.z = __expf((x.z - c.z) * invt - m) * invZ;
        o.w = __expf((x.w - c.w) * invt - m) * invZ;
        q4[d] = o;
    }
}

// ---------------------------------------------------------------- K2
// tstats[row] = { max_d y, sum_d exp(y - max) },  y = (t_reg - cgrid)/temp
__global__ void k_teacher_region_stats(const float* __restrict__ treg,
                                       const float* __restrict__ cgrid,
                                       const int* __restrict__ epoch,
                                       float2* __restrict__ tstats) {
    __shared__ float sm[256];
    const int row = blockIdx.x;
    const float invt = 1.0f / temp_from_epoch(*epoch);
    const float4* x4 = (const float4*)(treg + (size_t)row * DD);
    const float4* c4 = (const float4*)cgrid;

    float m = NEGINF;
    for (int d = threadIdx.x; d < DD / 4; d += blockDim.x) {
        float4 x = x4[d], c = c4[d];
        m = fmaxf(m, fmaxf(fmaxf((x.x - c.x) * invt, (x.y - c.y) * invt),
                           fmaxf((x.z - c.z) * invt, (x.w - c.w) * invt)));
    }
    m = block_max(m, sm);
    float s = 0.0f;
    for (int d = threadIdx.x; d < DD / 4; d += blockDim.x) {
        float4 x = x4[d], c = c4[d];
        s += __expf((x.x - c.x) * invt - m) + __expf((x.y - c.y) * invt - m)
           + __expf((x.z - c.z) * invt - m) + __expf((x.w - c.w) * invt - m);
    }
    s = block_sum(s, sm);
    if (threadIdx.x == 0) tstats[row] = make_float2(m, s);
}

// ---------------------------------------------------------------- K3
// out[row] = logsumexp(x[row] * scale)   (float4 streaming)
__global__ void k_row_lse(const float* __restrict__ x, float scale,
                          float* __restrict__ out) {
    __shared__ float sm[256];
    const int row = blockIdx.x;
    const float4* p4 = (const float4*)(x + (size_t)row * DD);

    float m = NEGINF;
    for (int d = threadIdx.x; d < DD / 4; d += blockDim.x) {
        float4 p = p4[d];
        m = fmaxf(m, fmaxf(fmaxf(p.x, p.y), fmaxf(p.z, p.w)) * scale);
    }
    m = block_max(m, sm);
    float s = 0.0f;
    for (int d = threadIdx.x; d < DD / 4; d += blockDim.x) {
        float4 p = p4[d];
        s += __expf(p.x * scale - m) + __expf(p.y * scale - m)
           + __expf(p.z * scale - m) + __expf(p.w * scale - m);
    }
    s = block_sum(s, sm);
    if (threadIdx.x == 0) out[row] = m + __logf(s);
}

// ---------------------------------------------------------------- K4
// out[row] = f16(x[row] / max(||x[row]||, 1e-12))
__global__ void k_l2norm_f16(const float* __restrict__ in,
                             _Float16* __restrict__ out) {
    __shared__ float sm[256];
    const int row = blockIdx.x;
    const float4* p4 = (const float4*)(in + (size_t)row * FF);
    float4 p = make_float4(0.f, 0.f, 0.f, 0.f);
    float ss = 0.0f;
    if (threadIdx.x < FF / 4) {
        p = p4[threadIdx.x];
        ss = p.x * p.x + p.y * p.y + p.z * p.z + p.w * p.w;
    }
    ss = block_sum(ss, sm);
    const float inv = 1.0f / fmaxf(sqrtf(ss), 1e-12f);
    if (threadIdx.x < FF / 4) {
        v4h h;
        h[0] = (_Float16)(p.x * inv);
        h[1] = (_Float16)(p.y * inv);
        h[2] = (_Float16)(p.z * inv);
        h[3] = (_Float16)(p.w * inv);
        ((v4h*)(out + (size_t)row * FF))[threadIdx.x] = h;
    }
}

// ---------------------------------------------------------------- K5
// One wave per (s-tile, b). sim[s, 0..N) = A(16xF) * B(FxN) via
// v_wmma_f32_16x16x32_f16, 12 k-steps per n-tile, B double-buffered.
// A tile stays resident in 96 VGPRs. Argmax: per-lane running (max, tile)
// in registers (each lane owns column slot n%16 == lane%16); one cross-lane
// butterfly at the end.
__device__ __forceinline__ v16h load_b_tile(const _Float16* brow, int kk, int half) {
    const int kb = kk * 32 + half * 8;
    v8h lo = *(const v8h*)(brow + kb);
    v8h hi = *(const v8h*)(brow + kb + 16);
    v16h r;
#pragma unroll
    for (int e = 0; e < 8; ++e) { r[e] = lo[e]; r[8 + e] = hi[e]; }
    return r;
}

__global__ __launch_bounds__(32)
void k_sim_argmax(const _Float16* __restrict__ sfea,   // s_rows x FF, f16
                  const _Float16* __restrict__ tfea,   // t_rows x FF, f16
                  int s_chunk_base, int t_chunk_base,
                  int S, int N,
                  int* __restrict__ inds, int ind_base) {
    const int b     = blockIdx.y;
    const int stile = blockIdx.x;
    const int lane  = threadIdx.x;
    const int half  = lane >> 4;     // 0: lanes 0-15, 1: lanes 16-31
    const int m     = lane & 15;
    const int s0    = stile * 16;

    // --- A tile: row M = m, K groups of 8 selected by half (ISA 16-bit
    //     A-matrix 16x32 layout). Two b128 loads per 32-K step.
    int srow = s_chunk_base + b * S + min(s0 + m, S - 1);
    const _Float16* arow = sfea + (size_t)srow * FF;
    v16h A[FF / 32];
#pragma unroll
    for (int kk = 0; kk < FF / 32; ++kk) {
        const int kb = kk * 32 + half * 8;
        v8h lo = *(const v8h*)(arow + kb);
        v8h hi = *(const v8h*)(arow + kb + 16);
#pragma unroll
        for (int e = 0; e < 8; ++e) { A[kk][e] = lo[e]; A[kk][8 + e] = hi[e]; }
    }

    // per-lane running best over this lane's column slot (n = nt*16 + m)
    float bv[8]; int bnt[8];
#pragma unroll
    for (int v = 0; v < 8; ++v) { bv[v] = NEGINF; bnt[v] = 0; }

    const int ntiles = (N + 15) >> 4;
    for (int nt = 0; nt < ntiles; ++nt) {
        const int ncol = nt * 16 + m;   // this lane's B column
        const int trow = t_chunk_base + b * N + min(ncol, N - 1);
        const _Float16* brow = tfea + (size_t)trow * FF;

        v8f c = {};
        v16h B0 = load_b_tile(brow, 0, half);
#pragma unroll
        for (int kk = 0; kk < FF / 32 - 1; ++kk) {
            v16h B1 = load_b_tile(brow, kk + 1, half);  // overlaps WMMA below
            c = __builtin_amdgcn_wmma_f32_16x16x32_f16(
                    false, A[kk], false, B0, (short)0, c, false, false);
            B0 = B1;
        }
        c = __builtin_amdgcn_wmma_f32_16x16x32_f16(
                false, A[FF / 32 - 1], false, B0, (short)0, c, false, false);

        // per-lane tracking: 8 compares, no cross-lane traffic in the loop.
        const bool colok = (ncol < N);
#pragma unroll
        for (int v = 0; v < 8; ++v) {
            float val = colok ? c[v] : NEGINF;
            if (val > bv[v]) { bv[v] = val; bnt[v] = nt; }   // first-max in lane
        }
    }

    // single cross-lane argmax per row (rows v + half*8); butterfly over
    // masks 1,2,4,8 stays within each 16-lane half.
#pragma unroll
    for (int v = 0; v < 8; ++v) {
        float val = bv[v];
        int   idx = bnt[v] * 16 + m;
#pragma unroll
        for (int mask = 1; mask < 16; mask <<= 1) {
            float ov = __shfl_xor(val, mask, 32);
            int   oi = __shfl_xor(idx, mask, 32);
            if (ov > val || (ov == val && oi < idx)) { val = ov; idx = oi; }
        }
        if (m == 0) {                      // lanes 0 / 16 hold rows v / v+8
            const int s = s0 + half * 8 + v;
            if (s < S)
                inds[ind_base + b * S + s] = t_chunk_base + b * N + idx;
        }
    }
}

// ---------------------------------------------------------------- K6a
// region CE for one (b, s): ce = lse_s - dot(softmax_teacher_row, s_row*10)
// teacher softmax reconstructed on the fly from (max, Z) stats. float4 path.
__global__ void k_region_ce(const float* __restrict__ sreg,
                            const float* __restrict__ treg,
                            const float* __restrict__ cgrid,
                            const float2* __restrict__ tstats,
                            const float* __restrict__ lse_reg,
                            const int* __restrict__ inds, int ind_base,
                            const int* __restrict__ epoch,
                            int s_chunk_base, int S,
                            float weight, float* __restrict__ acc) {
    __shared__ float sm[256];
    const int b = blockIdx.y, s = blockIdx.x;
    const int srow = s_chunk_base + b * S + s;
    const int trow = inds[ind_base + b * S + s];
    const float invt = 1.0f / temp_from_epoch(*epoch);
    const float2 st  = tstats[trow];
    const float tmax = st.x;
    const float invZ = 1.0f / st.y;
    const float4* xs4 = (const float4*)(sreg + (size_t)srow * DD);
    const float4* xt4 = (const float4*)(treg + (size_t)trow * DD);
    const float4* cg4 = (const float4*)cgrid;
    float part = 0.0f;
    for (int d = threadIdx.x; d < DD / 4; d += blockDim.x) {
        __builtin_prefetch(xs4 + d + 512, 0, 0);
        __builtin_prefetch(xt4 + d + 512, 0, 0);
        float4 t = xt4[d], c = cg4[d], v = xs4[d];
        part += __expf((t.x - c.x) * invt - tmax) * v.x
              + __expf((t.y - c.y) * invt - tmax) * v.y
              + __expf((t.z - c.z) * invt - tmax) * v.z
              + __expf((t.w - c.w) * invt - tmax) * v.w;
    }
    float dot = block_sum(part, sm) * invZ * 10.0f;
    if (threadIdx.x == 0)
        atomicAdd(acc, weight * (lse_reg[srow] - dot));
}

// ---------------------------------------------------------------- K6b
// cls CE for one b: ce = lse_cls[j*B+b] - dot(q[i*B+b], s_cls_row*10)
__global__ void k_cls_ce(const float* __restrict__ scls,
                         const float* __restrict__ q,
                         const float* __restrict__ lse_cls,
                         int j, int iq, int Bb,
                         float weight, float* __restrict__ acc) {
    __shared__ float sm[256];
    const int b = blockIdx.x;
    const int srow = j * Bb + b;
    const float4* v4 = (const float4*)(scls + (size_t)srow * DD);
    const float4* q4 = (const float4*)(q + (size_t)(iq * Bb + b) * DD);
    float part = 0.0f;
    for (int d = threadIdx.x; d < DD / 4; d += blockDim.x) {
        float4 v = v4[d], qq = q4[d];
        part += qq.x * v.x + qq.y * v.y + qq.z * v.z + qq.w * v.w;
    }
    float dot = block_sum(part, sm) * 10.0f;
    if (threadIdx.x == 0)
        atomicAdd(acc, weight * (lse_cls[srow] - dot));
}

// ---------------------------------------------------------------------------
static inline void* ws_take(char* base, size_t& off, size_t bytes) {
    void* p = base + off;
    off = (off + bytes + 255) & ~(size_t)255;
    return p;
}

extern "C" void kernel_launch(void* const* d_in, const int* in_sizes, int n_in,
                              void* d_out, int out_size, void* d_ws, size_t ws_size,
                              hipStream_t stream) {
    (void)n_in; (void)out_size; (void)ws_size;
    const float* s_cls  = (const float*)d_in[0];
    const float* s_reg  = (const float*)d_in[1];
    const float* s_fea  = (const float*)d_in[2];
    const float* t_cls  = (const float*)d_in[3];
    const float* t_reg  = (const float*)d_in[4];
    const float* t_fea  = (const float*)d_in[5];
    const float* center = (const float*)d_in[6];
    const float* cgrid  = (const float*)d_in[7];
    const int*   epoch  = (const int*)d_in[8];

    // derive shapes from in_sizes (n_global/n_local live on device; shapes
    // are recoverable from element counts)
    const int ncb    = in_sizes[0] / DD;          // NCROPS * B
    const int s_rows = in_sizes[1] / DD;
    const int t_rows = in_sizes[4] / DD;
    const int B2     = in_sizes[3] / DD;          // 2 * B
    const int Bb     = B2 / 2;
    const int NG     = t_rows / B2;               // n_global
    const int NC     = ncb / Bb;                  // NCROPS
    const int NL     = (s_rows / Bb - 2 * NG) / (NC - 2);
    const int nterms = 2 * (NC - 1);

    // workspace carve-up
    size_t off = 0;
    char* w = (char*)d_ws;
    float*     acc     = (float*)    ws_take(w, off, sizeof(float));
    float*     q       = (float*)    ws_take(w, off, (size_t)B2 * DD * 4);
    float2*    tstats  = (float2*)   ws_take(w, off, (size_t)t_rows * 8);
    float*     lse_reg = (float*)    ws_take(w, off, (size_t)s_rows * 4);
    float*     lse_cls = (float*)    ws_take(w, off, (size_t)ncb * 4);
    _Float16*  sfea_h  = (_Float16*) ws_take(w, off, (size_t)s_rows * FF * 2);
    _Float16*  tfea_h  = (_Float16*) ws_take(w, off, (size_t)t_rows * FF * 2);
    const int total_inds = 2 * Bb * (NG + (NC - 2) * NL);
    int*       inds    = (int*)      ws_take(w, off, (size_t)total_inds * 4);

    k_zero<<<1, 1, 0, stream>>>(acc);
    k_teacher_cls_softmax<<<B2, 256, 0, stream>>>(t_cls, center, epoch, q);
    k_teacher_region_stats<<<t_rows, 256, 0, stream>>>(t_reg, cgrid, epoch, tstats);
    k_row_lse<<<s_rows, 256, 0, stream>>>(s_reg, 10.0f, lse_reg);
    k_row_lse<<<ncb,    256, 0, stream>>>(s_cls, 10.0f, lse_cls);
    k_l2norm_f16<<<s_rows, 128, 0, stream>>>(s_fea, sfea_h);
    k_l2norm_f16<<<t_rows, 128, 0, stream>>>(t_fea, tfea_h);

    // student region chunk row offsets
    int s_off[16];
    s_off[0] = 0;
    for (int j = 0; j < NC; ++j)
        s_off[j + 1] = s_off[j] + ((j < 2) ? NG : NL) * Bb;

    int ind_off = 0;
    for (int i = 0; i < 2; ++i) {
        const int t_base = i * Bb * NG;
        for (int j = 0; j < NC; ++j) {
            if (j == i) continue;
            const int S = (j < 2) ? NG : NL;
            const int sbase = s_off[j];

            dim3 gsim((S + 15) / 16, Bb);
            k_sim_argmax<<<gsim, 32, 0, stream>>>(
                sfea_h, tfea_h, sbase, t_base, S, NG, inds, ind_off);

            const float wreg = 0.5f / ((float)S * (float)Bb * (float)nterms);
            dim3 gce(S, Bb);
            k_region_ce<<<gce, 256, 0, stream>>>(
                s_reg, t_reg, cgrid, tstats, lse_reg,
                inds, ind_off, epoch, sbase, S, wreg, acc);

            const float wcls = 0.5f / ((float)Bb * (float)nterms);
            k_cls_ce<<<Bb, 256, 0, stream>>>(
                s_cls, q, lse_cls, j, i, Bb, wcls, acc);

            ind_off += Bb * S;
        }
    }

    k_finalize<<<1, 1, 0, stream>>>(acc, (float*)d_out);
}
